// SelfLayer_50259707298436
// MI455X (gfx1250) — matrix operand order repaired and checked by reading, not compiled, not run on hardware
//
#include <hip/hip_runtime.h>

typedef __attribute__((ext_vector_type(16))) __bf16 v16bf;
typedef __attribute__((ext_vector_type(8)))  float  v8f;
typedef unsigned short u16;

#define B_  16
#define S_  1024
#define E_  1024
#define KS_ 128

// assemble a v16bf A/B fragment from two 16-byte (8 x bf16) chunks
static __device__ __forceinline__ v16bf pack2(uint4 lo, uint4 hi) {
    union { uint4 u[2]; v16bf v; } t;
    t.u[0] = lo; t.u[1] = hi;
    return t.v;
}

// ---------------------------------------------------------------------------
// W transpose+convert: Wt[n][e] = bf16(W[e][n]).  256 KB/matrix, L2-resident;
// makes every projection B-fragment two contiguous 16B loads.
// ---------------------------------------------------------------------------
__global__ __launch_bounds__(128)
void wconv_kernel(const float* __restrict__ W, __bf16* __restrict__ Wt) {
    const int e = blockIdx.x;       // 0..1023
    const int n = threadIdx.x;      // 0..127 (coalesced read of one W row)
    Wt[(size_t)n * E_ + e] = (__bf16)W[(size_t)e * KS_ + n];
}

// ---------------------------------------------------------------------------
// Projection: one wave computes a full 16x128 row panel (8 accumulators),
// reading its 16 rows of X exactly once. 8 x 32 = 256 WMMAs per wave.
// transposed=1 stores V as [b][n][s] bf16 (contiguous b128 stores, and makes
// the attention P*V B-fragments contiguous).  Q/K stores are coalesced via an
// LDS tile (b128 global stores).
// ---------------------------------------------------------------------------
__global__ __launch_bounds__(128)
void proj_kernel(const float* __restrict__ X, const __bf16* __restrict__ Wt,
                 u16* __restrict__ Out, int transposed) {
    __shared__ __bf16 ctile[4][16][128];             // 16 KB: per-wave store staging

    const int lane  = threadIdx.x & 31;
    const int wid   = threadIdx.x >> 5;
    const int hi    = lane >> 4;
    const int ln    = lane & 15;
    const int mbase = (blockIdx.x * 4 + wid) * 16;   // row tile in [0, B*S)

    v8f acc[8];
    #pragma unroll
    for (int nt = 0; nt < 8; ++nt) acc[nt] = (v8f){};

    const float* xrow = X + (size_t)(mbase + ln) * E_;   // A: lane holds row M=ln

    for (int e = 0; e < E_; e += 32) {
        // A fragment 16x32: comps 0-15 = K[k0..k0+15]
        const int k0 = e + hi * 16;
        float4 a0 = *(const float4*)(xrow + k0);
        float4 a1 = *(const float4*)(xrow + k0 + 4);
        float4 a2 = *(const float4*)(xrow + k0 + 8);
        float4 a3 = *(const float4*)(xrow + k0 + 12);
        v16bf af;
        af[0]  = (__bf16)a0.x; af[1]  = (__bf16)a0.y; af[2]  = (__bf16)a0.z; af[3]  = (__bf16)a0.w;
        af[4]  = (__bf16)a1.x; af[5]  = (__bf16)a1.y; af[6]  = (__bf16)a1.z; af[7]  = (__bf16)a1.w;
        af[8]  = (__bf16)a2.x; af[9]  = (__bf16)a2.y; af[10] = (__bf16)a2.z; af[11] = (__bf16)a2.w;
        af[12] = (__bf16)a3.x; af[13] = (__bf16)a3.y; af[14] = (__bf16)a3.z; af[15] = (__bf16)a3.w;

        // B fragments: lane = col n, comps = e rows -> contiguous in Wt[n][e]
        #pragma unroll
        for (int nt = 0; nt < 8; ++nt) {
            const __bf16* wrow = Wt + (size_t)(nt * 16 + ln) * E_ + e + hi * 16;
            v16bf bw = pack2(*(const uint4*)wrow, *(const uint4*)(wrow + 8));
            acc[nt] = __builtin_amdgcn_wmma_f32_16x16x32_bf16(
                          false, af, false, bw, (short)0, acc[nt], false, false);
        }
    }

    if (transposed) {
        // V^T [b][n][s]: lane's 8 rows contiguous in s -> one b128 store per nt
        const int b  = mbase >> 10;
        const int sb = (mbase & (S_ - 1)) + hi * 8;
        #pragma unroll
        for (int nt = 0; nt < 8; ++nt) {
            const int n = nt * 16 + ln;
            union { __bf16 b[8]; uint4 u; } o;
            #pragma unroll
            for (int r = 0; r < 8; ++r) o.b[r] = (__bf16)acc[nt][r];
            *(uint4*)(Out + (((size_t)b * KS_ + n) << 10) + sb) = o.u;
        }
    } else {
        // stage C-layout tile in LDS, then coalesced b128 row-major stores
        #pragma unroll
        for (int nt = 0; nt < 8; ++nt)
            #pragma unroll
            for (int r = 0; r < 8; ++r)
                ctile[wid][r + hi * 8][nt * 16 + ln] = (__bf16)acc[nt][r];
        // same wave wrote the tile; LDS ops are in-order within a wave
        const uint4* src = (const uint4*)&ctile[wid][lane >> 1][(lane & 1) * 64];
        uint4* dst = (uint4*)(Out + (size_t)(mbase + (lane >> 1)) * KS_ + (lane & 1) * 64);
        #pragma unroll
        for (int i = 0; i < 8; ++i) dst[i] = src[i];
    }
}

// ---------------------------------------------------------------------------
// Attention: 2 waves per (b, 16-row q tile), each covering half of s (512).
// Flash-style, no max-subtraction (matches reference's raw exp): accumulate
// unnormalized O and rowsum, combine the two s-halves through LDS, divide once.
// ---------------------------------------------------------------------------
__global__ __launch_bounds__(128)
void attn_kernel(const u16* __restrict__ Qb, const u16* __restrict__ Kb,
                 const u16* __restrict__ Vt, const float* __restrict__ mask,
                 float* __restrict__ Out) {
    __shared__ __bf16 pt[4][16][32];        // per-wave P tile (C-layout -> A-layout)
    __shared__ float  obuf[2][16][128];     // s-half partial O
    __shared__ float  rsbuf[2][16];         // s-half partial rowsum

    const int lane  = threadIdx.x & 31;
    const int wid   = threadIdx.x >> 5;
    const int hi    = lane >> 4;
    const int ln    = lane & 15;
    const int qsel  = wid >> 1;             // which q tile within block
    const int shalf = wid & 1;              // which s half
    const int widx  = blockIdx.x * 2 + qsel;          // 0..1023
    const int b     = widx >> 6;
    const int qbase = (widx & 63) * 16;

    // Q fragments (K-dim 0..127, 4 chunks of 32), contiguous b128 loads
    const u16* qrow = Qb + (size_t)(b * S_ + qbase + ln) * KS_;
    v16bf aq[4];
    #pragma unroll
    for (int c = 0; c < 4; ++c) {
        const int k0 = c * 32 + hi * 16;
        aq[c] = pack2(*(const uint4*)(qrow + k0), *(const uint4*)(qrow + k0 + 8));
    }

    v8f o[8];
    #pragma unroll
    for (int nt = 0; nt < 8; ++nt) o[nt] = (v8f){};
    float rs[8];
    #pragma unroll
    for (int r = 0; r < 8; ++r) rs[r] = 0.f;

    const float scale = 0.03125f;  // 1/sqrt(1024)
    const u16*  Kbb  = Kb + (size_t)b * S_ * KS_;
    const u16*  Vbb  = Vt + ((size_t)b * KS_ << 10);
    const float* mrow = mask + ((size_t)b * S_ + qbase) * S_;

    const int send = shalf * 512 + 512;
    for (int st = shalf * 512; st < send; st += 32) {
        // ---- scores: two 16x16 tiles, 4 WMMAs (K=32) each ----
        #pragma unroll
        for (int j = 0; j < 2; ++j) {
            v8f sc = (v8f){};
            const u16* krow = Kbb + (size_t)(st + j * 16 + ln) * KS_ + hi * 16;
            #pragma unroll
            for (int c = 0; c < 4; ++c) {
                v16bf bk = pack2(*(const uint4*)(krow + c * 32),
                                 *(const uint4*)(krow + c * 32 + 8));
                sc = __builtin_amdgcn_wmma_f32_16x16x32_bf16(
                         false, aq[c], false, bk, (short)0, sc, false, false);
            }
            // exp * mask, rowsum accumulate, stash P (bf16) in LDS
            const int scol = st + j * 16 + ln;
            #pragma unroll
            for (int r = 0; r < 8; ++r) {
                float m = mrow[(size_t)(r + hi * 8) * S_ + scol];
                float p = __expf(sc[r] * scale) * m;
                rs[r] += p;
                pt[wid][r + hi * 8][j * 16 + ln] = (__bf16)p;
            }
        }

        // ---- P (16x32) as A fragment from LDS (2 x ds_load_b128) ----
        const __bf16* prow = &pt[wid][ln][0];
        v16bf pf = pack2(*(const uint4*)(prow + hi * 16),
                         *(const uint4*)(prow + hi * 16 + 8));

        // ---- O += P * V : 8 column tiles, contiguous V^T b128 loads ----
        #pragma unroll
        for (int nt = 0; nt < 8; ++nt) {
            const u16* vrow = Vbb + (((size_t)(nt * 16 + ln)) << 10) + st + hi * 16;
            v16bf bv = pack2(*(const uint4*)vrow, *(const uint4*)(vrow + 8));
            o[nt] = __builtin_amdgcn_wmma_f32_16x16x32_bf16(
                        false, pf, false, bv, (short)0, o[nt], false, false);
        }
    }

    // ---- in-wave rowsum reduction across the 16 lanes of each half ----
    #pragma unroll
    for (int r = 0; r < 8; ++r) {
        float v = rs[r];
        v += __shfl_xor(v, 1, 32);
        v += __shfl_xor(v, 2, 32);
        v += __shfl_xor(v, 4, 32);
        v += __shfl_xor(v, 8, 32);
        rs[r] = v;
    }

    // ---- combine the two s-halves ----
    if (shalf == 1) {
        #pragma unroll
        for (int nt = 0; nt < 8; ++nt)
            #pragma unroll
            for (int r = 0; r < 8; ++r)
                obuf[qsel][r + hi * 8][nt * 16 + ln] = o[nt][r];
        if (ln == 0) {
            #pragma unroll
            for (int r = 0; r < 8; ++r) rsbuf[qsel][r + hi * 8] = rs[r];
        }
    }
    __syncthreads();
    if (shalf == 0) {
        float inv[8];
        #pragma unroll
        for (int r = 0; r < 8; ++r)
            inv[r] = 1.0f / (rs[r] + rsbuf[qsel][r + hi * 8]);
        float* orow = Out + (size_t)(b * S_ + qbase) * KS_;
        #pragma unroll
        for (int nt = 0; nt < 8; ++nt)
            #pragma unroll
            for (int r = 0; r < 8; ++r)
                orow[(size_t)(r + hi * 8) * KS_ + nt * 16 + ln] =
                    (o[nt][r] + obuf[qsel][r + hi * 8][nt * 16 + ln]) * inv[r];
    }
}

// ---------------------------------------------------------------------------
extern "C" void kernel_launch(void* const* d_in, const int* in_sizes, int n_in,
                              void* d_out, int out_size, void* d_ws, size_t ws_size,
                              hipStream_t stream) {
    const float* x_q  = (const float*)d_in[0];
    const float* x_k  = (const float*)d_in[1];
    const float* x_v  = (const float*)d_in[2];
    const float* mask = (const float*)d_in[3];
    const float* Wq   = (const float*)d_in[4];
    const float* Wk   = (const float*)d_in[5];
    const float* Wv   = (const float*)d_in[6];

    u16* Qb = (u16*)d_ws;                         // [B*S,128] bf16 row-major
    u16* Kb = Qb + (size_t)B_ * S_ * KS_;         // [B*S,128] bf16 row-major
    u16* Vt = Kb + (size_t)B_ * S_ * KS_;         // [B,128,S] bf16 (transposed)
    __bf16* Wtq = (__bf16*)(Vt + (size_t)B_ * S_ * KS_);  // [128,1024] bf16 each
    __bf16* Wtk = Wtq + (size_t)KS_ * E_;
    __bf16* Wtv = Wtk + (size_t)KS_ * E_;

    wconv_kernel<<<dim3(E_), 128, 0, stream>>>(Wq, Wtq);
    wconv_kernel<<<dim3(E_), 128, 0, stream>>>(Wk, Wtk);
    wconv_kernel<<<dim3(E_), 128, 0, stream>>>(Wv, Wtv);

    dim3 pgrid(B_ * S_ / 16 / 4);                 // 256 blocks, 4 waves, 1 row-panel/wave
    proj_kernel<<<pgrid, 128, 0, stream>>>(x_q, Wtq, Qb, 0);
    proj_kernel<<<pgrid, 128, 0, stream>>>(x_k, Wtk, Kb, 0);
    proj_kernel<<<pgrid, 128, 0, stream>>>(x_v, Wtv, Vt, 1);

    attn_kernel<<<dim3(B_ * S_ / 16 / 2), 128, 0, stream>>>(Qb, Kb, Vt, mask,
                                                            (float*)d_out);
}
